// LaplaceEncoder_29386166239262
// MI455X (gfx1250) — matrix operand order repaired and checked by reading, not compiled, not run on hardware
//
#include <hip/hip_runtime.h>
#include <hip/hip_bf16.h>

#define BB 4
#define TC 1024
#define TQ 1024
#define HD 64
#define SPLIT 4                       // flash-decoding context splits
#define CT_PER_SPLIT (TC / 32 / SPLIT)
#define NLOG2E_HALF (-0.72134752044448169f)   // -0.5 * log2(e)

typedef __attribute__((ext_vector_type(16))) _Float16 v16h;
typedef __attribute__((ext_vector_type(8)))  _Float16 v8h;
typedef __attribute__((ext_vector_type(8)))  float    v8f;

// ---------------------------------------------------------------------------
// Encoder: value (f16, transposed [B,H,Tc]) + key (f32 [B,Tc,H]).
// b_ctx cancels in |key - query|, so it is omitted from key/query.
// ---------------------------------------------------------------------------
__global__ __launch_bounds__(256)
void lap_encode_ctx(const float* __restrict__ cx, const float* __restrict__ cy,
                    const float* __restrict__ W_in, const float* __restrict__ b_in,
                    const float* __restrict__ W_ctx,
                    float* __restrict__ keyg, _Float16* __restrict__ vtg) {
  int idx = blockIdx.x * 256 + threadIdx.x;       // b*65536 + c*64 + h
  int h = idx & 63;
  int c = (idx >> 6) & (TC - 1);
  int b = idx >> 16;
  int row = b * TC + c;
  float x0 = cx[row * 2 + 0];
  float x1 = cx[row * 2 + 1];
  float y  = cy[row];
  float val = x0 * W_in[h] + x1 * W_in[HD + h] + y * W_in[2 * HD + h] + b_in[h];
  float k   = x0 * W_ctx[h] + x1 * W_ctx[HD + h];
  keyg[idx] = k;
  vtg[((size_t)b * HD + h) * TC + c] = (_Float16)val;
}

__global__ __launch_bounds__(256)
void lap_encode_qry(const float* __restrict__ tx, const float* __restrict__ W_ctx,
                    float* __restrict__ qg) {
  int idx = blockIdx.x * 256 + threadIdx.x;       // b*65536 + q*64 + h
  int h = idx & 63;
  int q = (idx >> 6) & (TQ - 1);
  int b = idx >> 16;
  int row = b * TQ + q;
  float x0 = tx[row * 2 + 0];
  float x1 = tx[row * 2 + 1];
  qg[idx] = x0 * W_ctx[h] + x1 * W_ctx[HD + h];
}

// ---------------------------------------------------------------------------
// Split-K flash L1-attention partial pass.
// Grid: B * (TQ/64) * SPLIT blocks of 128 threads (4 waves).  Each wave owns
// 16 queries and 256 contexts (8 K=32 tiles).  Scores are kept in the log2
// domain so exp == single v_exp_f32.  Emits unnormalized acc + (m, l).
// Round-1 loop skeleton (single-buffer tiles, two barriers, JIT B loads):
// the only structure the allocator compiles without spills / vgpr-msb churn.
// ---------------------------------------------------------------------------
__global__ __launch_bounds__(128)
void lap_attn_partial(const float* __restrict__ keyg, const float* __restrict__ qg,
                      const _Float16* __restrict__ vtg,
                      float* __restrict__ part_acc,
                      float* __restrict__ part_m, float* __restrict__ part_l) {
  __shared__ float    s_key[32 * HD];        // key tile [c][h]       (8 KB)
  __shared__ _Float16 s_vT[HD * 40];         // value tile [h][c]+pad (5 KB)

  const int tid  = threadIdx.x;
  const int lane = tid & 31;
  const int wave = tid >> 5;
  const int n    = lane & 15;    // WMMA column / query row within tile
  const int g    = lane >> 4;    // lane half-group
  const int bid  = blockIdx.x;
  const int b    = bid >> 6;
  const int qblk = (bid >> 2) & 15;
  const int sp   = bid & (SPLIT - 1);
  const int qbase = qblk * 64 + wave * 16;

  // Query row q = qbase + n held in registers (both lane groups load same row).
  float qv[HD];
  {
    const float4* qp = (const float4*)(qg + ((size_t)b * TQ + qbase + n) * HD);
#pragma unroll
    for (int i = 0; i < 16; ++i) {
      float4 t = qp[i];
      qv[4 * i + 0] = t.x; qv[4 * i + 1] = t.y;
      qv[4 * i + 2] = t.z; qv[4 * i + 3] = t.w;
    }
  }

  v8f acc[4];
#pragma unroll
  for (int ch = 0; ch < 4; ++ch)
#pragma unroll
    for (int r = 0; r < 8; ++r) acc[ch][r] = 0.0f;

  float m_run = -INFINITY;   // log2-domain running max
  float l_run = 0.0f;

  const int ct_end = (sp + 1) * CT_PER_SPLIT;
  for (int ct = sp * CT_PER_SPLIT; ct < ct_end; ++ct) {
    const int c0 = ct * 32;
    __syncthreads();
    // Stage key tile: 32x64 f32, fully coalesced.
    {
      const float4* kg4 = (const float4*)(keyg + ((size_t)b * TC + c0) * HD);
      float4* kl = (float4*)s_key;
#pragma unroll
      for (int i = 0; i < 4; ++i) kl[tid + i * 128] = kg4[tid + i * 128];
    }
    // Stage value tile transposed: s_vT[h][c - c0], row stride 40 halves.
    {
      int h = tid >> 1, half = tid & 1;
      const uint4* src = (const uint4*)(vtg + ((size_t)b * HD + h) * TC + c0 + half * 16);
      uint4* dst = (uint4*)(s_vT + h * 40 + half * 16);
      dst[0] = src[0];
      dst[1] = src[1];
    }
    __syncthreads();

    // 16 L1 scores per lane, placed in the 16-bit A-matrix K order:
    // lane group g: slots e<8 -> K = 8g + e ; e>=8 -> K = 8 + 8g + e.
    float p[16];
#pragma unroll
    for (int e = 0; e < 16; ++e) {
      const int k = (e < 8) ? (8 * g + e) : (8 + 8 * g + e);
      const float4* kp = (const float4*)(s_key + k * HD);  // group broadcast
      float s0 = 0.0f, s1 = 0.0f;
#pragma unroll
      for (int hh = 0; hh < 16; ++hh) {
        float4 kvv = kp[hh];
        s0 += fabsf(kvv.x - qv[4 * hh + 0]);
        s1 += fabsf(kvv.y - qv[4 * hh + 1]);
        s0 += fabsf(kvv.z - qv[4 * hh + 2]);
        s1 += fabsf(kvv.w - qv[4 * hh + 3]);
      }
      p[e] = NLOG2E_HALF * (s0 + s1);   // log2-domain score
    }

    // Online softmax (row q = n lives in lanes n and n+16).  Tree reductions.
    float mm[4];
#pragma unroll
    for (int e = 0; e < 4; ++e) mm[e] = p[e];
#pragma unroll
    for (int e = 4; e < 16; ++e) mm[e & 3] = fmaxf(mm[e & 3], p[e]);
    float mt = fmaxf(fmaxf(mm[0], mm[1]), fmaxf(mm[2], mm[3]));
    mt = fmaxf(mt, __shfl_xor(mt, 16));
    float mn = fmaxf(m_run, mt);
    float alpha = __builtin_amdgcn_exp2f(m_run - mn);
    float ss[4] = {0.0f, 0.0f, 0.0f, 0.0f};
#pragma unroll
    for (int e = 0; e < 16; ++e) {
      p[e] = __builtin_amdgcn_exp2f(p[e] - mn);
      ss[e & 3] += p[e];
    }
    float ssum = (ss[0] + ss[1]) + (ss[2] + ss[3]);
    ssum += __shfl_xor(ssum, 16);
    l_run = l_run * alpha + ssum;
    m_run = mn;

    // Rescale accumulators: D-layout row of VGPR r is (8g + r).
#pragma unroll
    for (int r = 0; r < 8; ++r) {
      float ar = __shfl(alpha, 8 * g + r);
#pragma unroll
      for (int ch = 0; ch < 4; ++ch) acc[ch][r] *= ar;
    }

    // Pack P into the f16 A operand.
    v16h a;
#pragma unroll
    for (int e = 0; e < 16; ++e) a[e] = (_Float16)p[e];

    // B operands just-in-time: transposed value tile, K in table order (8g.. and
    // 16+8g..), two aligned 16B LDS loads per chunk; 4 WMMAs back-to-back.
    const v8h* vt8 = (const v8h*)s_vT;
#pragma unroll
    for (int ch = 0; ch < 4; ++ch) {
      const int h0 = ch * 16;
      v8h lo = vt8[(h0 + n) * 5 + g];
      v8h hi = vt8[(h0 + n) * 5 + 2 + g];
      v16h bm = __builtin_shufflevector(lo, hi, 0, 1, 2, 3, 4, 5, 6, 7,
                                        8, 9, 10, 11, 12, 13, 14, 15);
      acc[ch] = __builtin_amdgcn_wmma_f32_16x16x32_f16(
          false, a, false, bm, (short)0, acc[ch], false, false);
    }
  }

  // Emit unnormalized partials.
  const size_t rowbase = (size_t)(sp * BB + b) * TQ + qbase;
#pragma unroll
  for (int r = 0; r < 8; ++r)
#pragma unroll
    for (int ch = 0; ch < 4; ++ch)
      part_acc[(rowbase + 8 * g + r) * HD + ch * 16 + n] = acc[ch][r];
  if (g == 0) {
    part_m[rowbase + n] = m_run;
    part_l[rowbase + n] = l_run;
  }
}

// ---------------------------------------------------------------------------
// Combine partials + output projection rep @ W_tgt + b_tgt.
// Grid: B * (TQ/64) blocks of 128 threads; lane -> row q = n, cols g*32..+31.
// ---------------------------------------------------------------------------
__global__ __launch_bounds__(128)
void lap_combine(const float* __restrict__ part_acc,
                 const float* __restrict__ part_m, const float* __restrict__ part_l,
                 const float* __restrict__ Wt, const float* __restrict__ bt,
                 float* __restrict__ outg) {
  __shared__ float s_rep[4 * 16 * 68];   // per-wave rep rows (stride 68 floats)
  __shared__ float s_wt[HD * HD];

  const int tid  = threadIdx.x;
  const int lane = tid & 31;
  const int wave = tid >> 5;
  const int n    = lane & 15;
  const int g    = lane >> 4;
  const int b    = blockIdx.x >> 4;
  const int qblk = blockIdx.x & 15;
  const int qbase = qblk * 64 + wave * 16;
  const int q = qbase + n;

  // Stage W_tgt.
  {
    const float4* wg = (const float4*)Wt;
    float4* wl = (float4*)s_wt;
#pragma unroll
    for (int i = 0; i < 8; ++i) wl[tid + i * 128] = wg[tid + i * 128];
  }

  float m[SPLIT], l[SPLIT];
#pragma unroll
  for (int s = 0; s < SPLIT; ++s) {
    size_t idx = (size_t)(s * BB + b) * TQ + q;
    m[s] = part_m[idx];
    l[s] = part_l[idx];
  }
  float M = fmaxf(fmaxf(m[0], m[1]), fmaxf(m[2], m[3]));
  float w[SPLIT];
  float lsum = 0.0f;
#pragma unroll
  for (int s = 0; s < SPLIT; ++s) {
    w[s] = __builtin_amdgcn_exp2f(m[s] - M);
    lsum += w[s] * l[s];
  }
  const float inv = __builtin_amdgcn_rcpf(lsum);

  float rep[32];
#pragma unroll
  for (int j = 0; j < 32; ++j) rep[j] = 0.0f;
#pragma unroll
  for (int s = 0; s < SPLIT; ++s) {
    const float4* ap = (const float4*)(part_acc +
        ((size_t)(s * BB + b) * TQ + q) * HD + g * 32);
    const float ws = w[s];
#pragma unroll
    for (int i = 0; i < 8; ++i) {
      float4 t = ap[i];
      rep[4 * i + 0] += ws * t.x;
      rep[4 * i + 1] += ws * t.y;
      rep[4 * i + 2] += ws * t.z;
      rep[4 * i + 3] += ws * t.w;
    }
  }
  {
    float* rrow = s_rep + (wave * 16 + n) * 68 + g * 32;
#pragma unroll
    for (int j = 0; j < 32; ++j) rrow[j] = rep[j] * inv;
  }
  __syncthreads();

  // out[q][ho] = b_tgt[ho] + sum_h rep[q][h] * Wt[h][ho]
  float o[32];
#pragma unroll
  for (int j = 0; j < 32; ++j) o[j] = bt[g * 32 + j];
  const float* reprow = s_rep + (wave * 16 + n) * 68;
  const float4* wt4 = (const float4*)s_wt;
#pragma unroll 4
  for (int h = 0; h < HD; ++h) {
    float rh = reprow[h];
#pragma unroll
    for (int i = 0; i < 8; ++i) {
      float4 wvv = wt4[h * 16 + g * 8 + i];   // broadcast within lane group
      o[4 * i + 0] += rh * wvv.x;
      o[4 * i + 1] += rh * wvv.y;
      o[4 * i + 2] += rh * wvv.z;
      o[4 * i + 3] += rh * wvv.w;
    }
  }
  float4* op = (float4*)(outg + (((size_t)b * TQ + qbase + n) * HD + g * 32));
#pragma unroll
  for (int i = 0; i < 8; ++i)
    op[i] = make_float4(o[4 * i + 0], o[4 * i + 1], o[4 * i + 2], o[4 * i + 3]);
}

// ---------------------------------------------------------------------------
extern "C" void kernel_launch(void* const* d_in, const int* in_sizes, int n_in,
                              void* d_out, int out_size, void* d_ws, size_t ws_size,
                              hipStream_t stream) {
  const float* cx    = (const float*)d_in[0];
  const float* cy    = (const float*)d_in[1];
  const float* tx    = (const float*)d_in[2];
  const float* W_in  = (const float*)d_in[3];
  const float* b_in  = (const float*)d_in[4];
  const float* W_ctx = (const float*)d_in[5];
  // d_in[6] = b_ctx: cancels in |key - query|, unused.
  const float* W_tgt = (const float*)d_in[7];
  const float* b_tgt = (const float*)d_in[8];
  float* out = (float*)d_out;

  char* ws = (char*)d_ws;
  size_t off = 0;
  float*    keyg = (float*)(ws + off);  off += (size_t)BB * TC * HD * 4;       // 1 MB
  float*    qg   = (float*)(ws + off);  off += (size_t)BB * TQ * HD * 4;       // 1 MB
  _Float16* vtg  = (_Float16*)(ws + off); off += (size_t)BB * TC * HD * 2;     // 0.5 MB
  float* part_acc = (float*)(ws + off); off += (size_t)SPLIT * BB * TQ * HD * 4; // 4 MB
  float* part_m   = (float*)(ws + off); off += (size_t)SPLIT * BB * TQ * 4;    // 64 KB
  float* part_l   = (float*)(ws + off);                                        // 64 KB

  lap_encode_ctx<<<BB * TC * HD / 256, 256, 0, stream>>>(cx, cy, W_in, b_in,
                                                         W_ctx, keyg, vtg);
  lap_encode_qry<<<BB * TQ * HD / 256, 256, 0, stream>>>(tx, W_ctx, qg);
  lap_attn_partial<<<BB * 16 * SPLIT, 128, 0, stream>>>(keyg, qg, vtg,
                                                        part_acc, part_m, part_l);
  lap_combine<<<BB * 16, 128, 0, stream>>>(part_acc, part_m, part_l,
                                           W_tgt, b_tgt, out);
}
